// QuantizedLinear_34823594836132
// MI455X (gfx1250) — compile-verified, matrix operand order checked by reading, not run on hardware
//
#include <hip/hip_runtime.h>
#include <hip/hip_bf16.h>
#include <stdint.h>

// ---------------------------------------------------------------------------
// Quantized linear (W8 / A-bf16) -> V_WMMA_F32_16X16X32_BF16 on gfx1250.
// y[m,n] = (sum_k x[m,k]*W[n,k]) * scale[n] + bias[n]
// M=8192, K=4096, N=11008 (all divisible by tile sizes, no bounds checks)
// ---------------------------------------------------------------------------

typedef __attribute__((ext_vector_type(16))) __bf16 v16bf;
typedef __attribute__((ext_vector_type(2)))  __bf16 v2bf;
typedef __attribute__((ext_vector_type(8)))  float  v8f;

#define M_DIM 8192
#define K_DIM 4096
#define N_DIM 11008

#define BM 128
#define BN 128
#define BK 32
#define LDSS 48               // row stride in bf16 elems: BK+16 pad -> 96B rows (16B aligned)
#define KTILES (K_DIM / BK)   // 128

// Pack two f32 into one dword of bf16 (RNE) using the HW converter.
__device__ __forceinline__ uint32_t pack2_bf16(float lo, float hi) {
    union { v2bf v; uint32_t u; } cv;
#if __has_builtin(__builtin_amdgcn_cvt_pk_bf16_f32)
    cv.v = __builtin_amdgcn_cvt_pk_bf16_f32(lo, hi);
#else
    cv.v[0] = (__bf16)lo;     // fptrunc = round-to-nearest-even
    cv.v[1] = (__bf16)hi;
#endif
    return cv.u;
}

union Frag16 { uint4 q[2]; v16bf v; };    // 32B fragment (8 VGPRs)

__global__ __launch_bounds__(256)
void qlinear_wmma_bf16(const float*  __restrict__ X,
                       const int8_t* __restrict__ Wq,
                       const float*  __restrict__ Wscale,
                       const float*  __restrict__ Bias,
                       float*        __restrict__ Y)
{
    __shared__ __align__(16) uint16_t As[2][BM * LDSS];  // 12KB per buffer
    __shared__ __align__(16) uint16_t Bs[2][BN * LDSS];  // 12KB per buffer

    const int tid  = threadIdx.x;
    const int lane = tid & 31;
    const int w    = tid >> 5;      // wave 0..7
    const int wm   = w >> 2;        // 0..1 : 64 M-rows per wave
    const int wn   = w & 3;         // 0..3 : 32 N-cols per wave
    const int half = lane >> 4;     // K-half select (contiguous k = 16*half + s, A and B alike)
    const int nl   = lane & 15;

    const int bm = blockIdx.y * BM;
    const int bn = blockIdx.x * BN;

    // ---- staging assignment: 2 threads per row, 16 contiguous K each ----
    const int srow = tid >> 1;            // 0..127
    const int scol = (tid & 1) << 4;      // 0 or 16

    const float*  gA = X  + (size_t)(bm + srow) * K_DIM + scol;
    const int8_t* gB = Wq + (size_t)(bn + srow) * K_DIM + scol;

    // ---- per-lane epilogue scale/bias (n = bn + wn*32 + nt*16 + nl) ----
    const int n0 = bn + wn * 32 + nl;
    const float sc0 = Wscale[n0];
    const float sc1 = Wscale[n0 + 16];
    const float bi0 = Bias[n0];
    const float bi1 = Bias[n0 + 16];

    v8f acc[4][2];
    #pragma unroll
    for (int i = 0; i < 4; ++i)
        #pragma unroll
        for (int j = 0; j < 2; ++j)
            acc[i][j] = (v8f){0.f,0.f,0.f,0.f,0.f,0.f,0.f,0.f};

    // ---- register-staged pipeline: fetch(t+1) overlaps compute(t) ----
    float4 ra[4];
    int4   rb;

    // prologue: fetch tile 0
    #pragma unroll
    for (int i = 0; i < 4; ++i) ra[i] = *(const float4*)(gA + i * 4);
    rb = *(const int4*)gB;

    for (int t = 0; t < KTILES; ++t) {
        const int buf = t & 1;

        // ---- convert staged regs -> LDS (bf16) via HW cvt ----
        {
            uint32_t p[8];
            #pragma unroll
            for (int i = 0; i < 4; ++i) {
                p[2*i]   = pack2_bf16(ra[i].x, ra[i].y);
                p[2*i+1] = pack2_bf16(ra[i].z, ra[i].w);
            }
            uint16_t* as = &As[buf][srow * LDSS + scol];
            *(uint4*)(as)     = make_uint4(p[0], p[1], p[2], p[3]);
            *(uint4*)(as + 8) = make_uint4(p[4], p[5], p[6], p[7]);

            const int dw[4] = { rb.x, rb.y, rb.z, rb.w };
            uint32_t q[8];
            #pragma unroll
            for (int d = 0; d < 4; ++d) {
                const int v = dw[d];
                q[2*d]   = pack2_bf16((float)(int8_t)(v),
                                      (float)(int8_t)(v >> 8));
                q[2*d+1] = pack2_bf16((float)(int8_t)(v >> 16),
                                      (float)(int8_t)(v >> 24));
            }
            uint16_t* bs = &Bs[buf][srow * LDSS + scol];
            *(uint4*)(bs)     = make_uint4(q[0], q[1], q[2], q[3]);
            *(uint4*)(bs + 8) = make_uint4(q[4], q[5], q[6], q[7]);
        }

        // Single barrier per iteration is sufficient: the next write to
        // buffer b happens two iterations after its last reader, and every
        // thread passes exactly one barrier in between (dscnt drained here).
        __syncthreads();

        // ---- issue global loads for next tile early (overlap with WMMA) ----
        if (t + 1 < KTILES) {
            const int ko = (t + 1) * BK;
            #pragma unroll
            for (int i = 0; i < 4; ++i) ra[i] = *(const float4*)(gA + ko + i * 4);
            rb = *(const int4*)(gB + ko);
        }

        // ---- WMMA compute on current buffer ----
        {
            Frag16 bfr[2];
            #pragma unroll
            for (int nt = 0; nt < 2; ++nt) {
                const uint16_t* p =
                    &Bs[buf][(wn * 32 + nt * 16 + nl) * LDSS + half * 16];
                bfr[nt].q[0] = *(const uint4*)(p);
                bfr[nt].q[1] = *(const uint4*)(p + 8);
            }
            #pragma unroll
            for (int mt = 0; mt < 4; ++mt) {
                const uint16_t* p =
                    &As[buf][(wm * 64 + mt * 16 + nl) * LDSS + half * 16];
                Frag16 af;
                af.q[0] = *(const uint4*)(p);
                af.q[1] = *(const uint4*)(p + 8);
                acc[mt][0] = __builtin_amdgcn_wmma_f32_16x16x32_bf16(
                    false, af.v, false, bfr[0].v, (short)0, acc[mt][0], false, false);
                acc[mt][1] = __builtin_amdgcn_wmma_f32_16x16x32_bf16(
                    false, af.v, false, bfr[1].v, (short)0, acc[mt][1], false, false);
            }
        }
    }

    // ---- epilogue: D layout -> vgpr r, lane L holds (m = r + half*8, n = nl) ----
    #pragma unroll
    for (int mt = 0; mt < 4; ++mt) {
        const size_t gm = (size_t)(bm + wm * 64 + mt * 16 + half * 8);
        #pragma unroll
        for (int nt = 0; nt < 2; ++nt) {
            const float s = nt ? sc1 : sc0;
            const float b = nt ? bi1 : bi0;
            float* o = Y + gm * N_DIM + (bn + wn * 32 + nt * 16 + nl);
            #pragma unroll
            for (int r = 0; r < 8; ++r)
                o[(size_t)r * N_DIM] = acc[mt][nt][r] * s + b;
        }
    }
}

extern "C" void kernel_launch(void* const* d_in, const int* in_sizes, int n_in,
                              void* d_out, int out_size, void* d_ws, size_t ws_size,
                              hipStream_t stream) {
    const float*  X  = (const float*)d_in[0];     // [4,2048,4096] f32
    const int8_t* Wq = (const int8_t*)d_in[1];    // [11008,4096] int8
    const float*  Ws = (const float*)d_in[2];     // [11008] f32
    const float*  Bi = (const float*)d_in[3];     // [11008] f32
    float*        Y  = (float*)d_out;             // [4,2048,11008] f32

    dim3 grid(N_DIM / BN, M_DIM / BM);            // (86, 64)
    qlinear_wmma_bf16<<<grid, 256, 0, stream>>>(X, Wq, Ws, Bi, Y);
}
